// MoENormalizationLayer_5506148073492
// MI455X (gfx1250) — matrix operand order repaired and checked by reading, not compiled, not run on hardware
//
#include <hip/hip_runtime.h>
#include <hip/hip_bf16.h>

typedef __attribute__((ext_vector_type(16))) _Float16 v16h;
typedef __attribute__((ext_vector_type(8)))  _Float16 v8h;
typedef __attribute__((ext_vector_type(8)))  float    v8f;

#define B_   64
#define C_   512
#define HW_  1024
#define E_   8
#define G_   32
#define CPG  16      // channels per group = C_/G_
#define EPS_ 1e-5f

// Workspace layout (bytes):
//   [0          , 65536) : flat_f16  [B][C]   _Float16
//   [65536      , 73728) : gmean     [B][G]   float
//   [73728      , 81920) : grstd     [B][G]   float
//   [81920      , 82432) : coeff     [B][2]   float
//   [82432      , 82944) : topk      [B][2]   int

// ---------------------------------------------------------------------------
// Kernel 1: fused stats pass. One block per (b, group).
//   - per-channel mean  -> flat_f16 (router input, f16 for WMMA)
//   - group mean / rstd -> gmean / grstd
// One read of x total. 16 threads per channel, float4 loads, 256B coalesced.
// ---------------------------------------------------------------------------
__global__ __launch_bounds__(256) void moe_stats_kernel(
    const float* __restrict__ x,
    _Float16* __restrict__ flat16,
    float* __restrict__ gmean,
    float* __restrict__ grstd)
{
    const int blk = blockIdx.x;           // b*G + g
    const int b   = blk >> 5;
    const int g   = blk & 31;
    const int t   = threadIdx.x;
    const int ci  = t >> 4;               // channel within group, 0..15
    const int sub = t & 15;               // sub-lane within channel
    const int c   = g * CPG + ci;

    const float* xp = x + ((size_t)(b * C_ + c)) * HW_;

    float s = 0.f, q = 0.f;
    #pragma unroll
    for (int it = 0; it < 16; ++it) {
        float4 v = *(const float4*)(xp + it * 64 + sub * 4);
        s += v.x + v.y + v.z + v.w;
        q += v.x * v.x + v.y * v.y + v.z * v.z + v.w * v.w;
    }

    // reduce the 16 lanes of one channel (wave32: two channels per wave)
    #pragma unroll
    for (int off = 8; off >= 1; off >>= 1) {
        s += __shfl_xor(s, off, 16);
        q += __shfl_xor(q, off, 16);
    }

    __shared__ float ch_s[CPG];
    __shared__ float ch_q[CPG];
    if (sub == 0) {
        flat16[b * C_ + c] = (_Float16)(s * (1.0f / (float)HW_));
        ch_s[ci] = s;
        ch_q[ci] = q;
    }
    __syncthreads();

    if (t == 0) {
        float gs = 0.f, gq = 0.f;
        #pragma unroll
        for (int i = 0; i < CPG; ++i) { gs += ch_s[i]; gq += ch_q[i]; }
        const float inv_n = 1.0f / (float)(CPG * HW_);
        const float m   = gs * inv_n;
        const float var = gq * inv_n - m * m;
        gmean[blk] = m;
        grstd[blk] = rsqrtf(var + EPS_);
    }
}

// ---------------------------------------------------------------------------
// Kernel 2: router via WMMA. Single block, 4 waves (one 16-row M tile each).
//   logits[64x16] = flat_f16[64x512] @ router_w^T (padded 512x16), f32 acc.
//   Then softmax over E=8, top-2, prob-weighted coefficients.
// A fragment (16-bit A 16x32): lane<16 holds K=k0+0..7 & k0+16..23 of row
// (lane&15); lane>=16 holds K=k0+8..15 & k0+24..31  -> two aligned v8h loads.
// B fragment (16-bit B 32x16): lane holds column (lane&15), same K order as A:
// K=kb..kb+7 & kb+16..kb+23 with kb = k0 + (lane<16 ? 0 : 8).
// ---------------------------------------------------------------------------
__global__ __launch_bounds__(128) void moe_router_kernel(
    const _Float16* __restrict__ flat16,
    const float* __restrict__ router_w,   // [E][C]
    const float* __restrict__ router_b,   // [E]
    float* __restrict__ coeff,            // [B][2]
    int* __restrict__ topk)               // [B][2]
{
    const int t    = threadIdx.x;
    const int wave = t >> 5;          // M tile 0..3
    const int lane = t & 31;
    const int n    = lane & 15;       // A row within tile / B column
    const int hi   = lane >> 4;       // lane half selector

    const _Float16* arow = flat16 + (wave * 16 + n) * C_ + (hi ? 8 : 0);

    v8f acc = {};
    for (int k0 = 0; k0 < C_; k0 += 32) {
        // A fragment
        v8h a0 = *(const v8h*)(arow + k0);
        v8h a1 = *(const v8h*)(arow + k0 + 16);
        v16h a;
        #pragma unroll
        for (int i = 0; i < 8; ++i) { a[i] = a0[i]; a[8 + i] = a1[i]; }

        // B fragment: router_w^T column n, rows kb..kb+7 & kb+16..kb+23 (zero pad n>=E)
        const int kb = k0 + (hi ? 8 : 0);
        v16h bf;
        if (n < E_) {
            const float* wr = router_w + n * C_ + kb;
            #pragma unroll
            for (int i = 0; i < 16; ++i) bf[i] = (_Float16)wr[(i < 8) ? i : i + 8];
        } else {
            #pragma unroll
            for (int i = 0; i < 16; ++i) bf[i] = (_Float16)0.0f;
        }

        acc = __builtin_amdgcn_wmma_f32_16x16x32_f16(
            /*neg_a=*/false, a, /*neg_b=*/false, bf,
            /*c_mod=*/(short)0, acc, /*reuse_a=*/false, /*reuse_b=*/false);
    }

    // D layout: VGPR r -> M = r + (hi?8:0), N = lane&15
    __shared__ float lds_logits[B_][16];
    const int mbase = wave * 16 + (hi ? 8 : 0);
    #pragma unroll
    for (int r = 0; r < 8; ++r) lds_logits[mbase + r][n] = acc[r];
    __syncthreads();

    if (t < B_) {
        const int b = t;
        float lg[E_];
        float mx = -1e30f;
        #pragma unroll
        for (int e = 0; e < E_; ++e) {
            lg[e] = lds_logits[b][e] + router_b[e];
            mx = fmaxf(mx, lg[e]);
        }
        float sum = 0.f;
        #pragma unroll
        for (int e = 0; e < E_; ++e) { lg[e] = __expf(lg[e] - mx); sum += lg[e]; }
        (void)sum; // softmax scale cancels in coeff = vals / sum(vals)

        int e0 = 0; float v0 = lg[0];
        #pragma unroll
        for (int e = 1; e < E_; ++e) if (lg[e] > v0) { v0 = lg[e]; e0 = e; }
        int e1 = (e0 == 0) ? 1 : 0; float v1 = lg[e1];
        #pragma unroll
        for (int e = 0; e < E_; ++e) {
            if (e == e0) continue;
            if (lg[e] > v1) { v1 = lg[e]; e1 = e; }
        }
        const float inv_tot = 1.0f / (v0 + v1);
        coeff[b * 2 + 0] = v0 * inv_tot;
        coeff[b * 2 + 1] = v1 * inv_tot;
        topk[b * 2 + 0] = e0;
        topk[b * 2 + 1] = e1;
    }
}

// ---------------------------------------------------------------------------
// Kernel 3: normalize + fused per-sample affine. One block per (b, c).
//   y = x * a + d,  a = rstd * fw,  d = fb - mean * a
// Second read of x should hit L2 (x = 134MB < 192MB L2). float4 in/out.
// ---------------------------------------------------------------------------
__global__ __launch_bounds__(256) void moe_norm_kernel(
    const float* __restrict__ x,
    const float* __restrict__ ew,     // [E][C]
    const float* __restrict__ eb,     // [E][C]
    const float* __restrict__ sw,     // [C]
    const float* __restrict__ sb,     // [C]
    const float* __restrict__ gmean,  // [B][G]
    const float* __restrict__ grstd,  // [B][G]
    const float* __restrict__ coeff,  // [B][2]
    const int*   __restrict__ topk,   // [B][2]
    float* __restrict__ out)
{
    const int blk = blockIdx.x;          // b*C + c
    const int b = blk >> 9;
    const int c = blk & (C_ - 1);
    const int g = c >> 4;                // 16 channels per group

    // uniform (scalar) computation of fused affine for this (b, c)
    const float m  = gmean[b * G_ + g];
    const float rs = grstd[b * G_ + g];
    const int   e0 = topk[b * 2 + 0];
    const int   e1 = topk[b * 2 + 1];
    const float c0 = coeff[b * 2 + 0];
    const float c1 = coeff[b * 2 + 1];
    const float fw = c0 * ew[e0 * C_ + c] + c1 * ew[e1 * C_ + c] + sw[c];
    const float fb = c0 * eb[e0 * C_ + c] + c1 * eb[e1 * C_ + c] + sb[c];
    const float a = rs * fw;
    const float d = fb - m * a;

    const size_t base = (size_t)blk * HW_ + threadIdx.x * 4;
    float4 v = *(const float4*)(x + base);
    float4 o;
    o.x = v.x * a + d;
    o.y = v.y * a + d;
    o.z = v.z * a + d;
    o.w = v.w * a + d;
    *(float4*)(out + base) = o;
}

// ---------------------------------------------------------------------------
extern "C" void kernel_launch(void* const* d_in, const int* in_sizes, int n_in,
                              void* d_out, int out_size, void* d_ws, size_t ws_size,
                              hipStream_t stream) {
    const float* x        = (const float*)d_in[0];
    const float* ew       = (const float*)d_in[1];
    const float* eb       = (const float*)d_in[2];
    const float* sw       = (const float*)d_in[3];
    const float* sb       = (const float*)d_in[4];
    const float* router_w = (const float*)d_in[5];
    const float* router_b = (const float*)d_in[6];
    float* out = (float*)d_out;

    char* ws = (char*)d_ws;
    _Float16* flat16 = (_Float16*)(ws);
    float*    gmean  = (float*)(ws + 65536);
    float*    grstd  = (float*)(ws + 65536 + 8192);
    float*    coeff  = (float*)(ws + 65536 + 16384);
    int*      topk   = (int*)  (ws + 65536 + 16384 + 512);

    moe_stats_kernel<<<B_ * G_, 256, 0, stream>>>(x, flat16, gmean, grstd);
    moe_router_kernel<<<1, 128, 0, stream>>>(flat16, router_w, router_b, coeff, topk);
    moe_norm_kernel<<<B_ * C_, 256, 0, stream>>>(x, ew, eb, sw, sb,
                                                 gmean, grstd, coeff, topk, out);
}
